// RationalQuadratic_33749853012409
// MI455X (gfx1250) — compile-verified, matrix operand order checked by reading, not run on hardware
//
#include <hip/hip_runtime.h>
#include <hip/hip_bf16.h>

// ---------------------------------------------------------------------------
// RQS flow model on MI455X (gfx1250, wave32, WMMA).
//   GEMM1/2 -> bf16 WMMA kernels (bias+ReLU fused, bf16 out), 4Mx2N per wave.
//   GEMM3   -> fused with the rational-quadratic-spline epilogue so the
//              16384 x 12032 parameter tensor (788 MB) never touches HBM.
//              2Mx3N accumulators per wave halve B-fragment traffic per WMMA.
// Weights transposed to [N x K] bf16 (L2-resident; 192 MB L2 holds all of
// them), so every WMMA B fragment is two contiguous 16B global loads.
// ---------------------------------------------------------------------------

typedef __attribute__((ext_vector_type(16))) __bf16 v16bf;
typedef __attribute__((ext_vector_type(8)))  __bf16 v8bf;
typedef __attribute__((ext_vector_type(8)))  float  v8f;
typedef __attribute__((ext_vector_type(4)))  int    v4i;
typedef unsigned short u16;

#define TAILC     3.0f
#define MIN_VAL   0.001f
#define NBINS     16
#define MIN_TOTAL (NBINS * MIN_VAL)

// ---- async global->LDS staging (gfx1250); falls back to VGPR copies -------
#if defined(__has_builtin)
#  if __has_builtin(__builtin_amdgcn_global_load_async_to_lds_b128) && \
      __has_builtin(__builtin_amdgcn_s_wait_asynccnt)
#    define USE_ASYNC_LDS 1
#  endif
#endif
#ifndef USE_ASYNC_LDS
#  define USE_ASYNC_LDS 0
#endif

static __device__ __forceinline__ void copy16_g2l(const u16* gsrc, u16* ldst) {
#if USE_ASYNC_LDS
    __builtin_amdgcn_global_load_async_to_lds_b128(
        (__attribute__((address_space(1))) v4i*)gsrc,
        (__attribute__((address_space(3))) v4i*)ldst, 0, 0);
#else
    *(uint4*)ldst = *(const uint4*)gsrc;
#endif
}
static __device__ __forceinline__ void staging_fence() {
#if USE_ASYNC_LDS
    __builtin_amdgcn_s_wait_asynccnt(0);
#endif
}

static __device__ __forceinline__ u16 f32_to_bf16(float f) {
    unsigned u = __float_as_uint(f);
    unsigned rnd = 0x7FFFu + ((u >> 16) & 1u);   // round-to-nearest-even
    return (u16)((u + rnd) >> 16);
}

// Build a 16-element bf16 WMMA fragment from two contiguous 16B chunks.
// Per ISA 7.12.2 (16-bit operands, wave32): low lanes hold K {0..7,16..23},
// high lanes K {8..15,24..31}; caller passes p0 = base + k0 + 8*half.
static __device__ __forceinline__ v16bf make_frag(const u16* p0) {
    v8bf lo = *(const v8bf*)p0;
    v8bf hi = *(const v8bf*)(p0 + 16);
    v16bf f;
#pragma unroll
    for (int i = 0; i < 8; ++i) { f[i] = lo[i]; f[8 + i] = hi[i]; }
    return f;
}

// ------------------------- elementwise converters --------------------------

__global__ void cvt_f32_bf16(const float* __restrict__ in, u16* __restrict__ out, int n) {
    int i = blockIdx.x * 256 + threadIdx.x;
    if (i < n) out[i] = f32_to_bf16(in[i]);
}

// W [K x N] row-major f32  ->  WT [N x K] row-major bf16
__global__ void transpose_cvt(const float* __restrict__ W, u16* __restrict__ WT,
                              int K, int N) {
    int i = blockIdx.x * 256 + threadIdx.x;
    if (i < N * K) {
        int n = i / K, k = i - n * K;
        WT[i] = f32_to_bf16(W[(size_t)k * N + n]);
    }
}

// ---------------- GEMM: C = relu(A @ B + bias), bf16 in/out ----------------
// A [M x K] bf16 row-major, BT [N x K] bf16 row-major, bias f32[N].
// Block 256 thr = 8 waves; tile 128x128; wave = 4 row-tiles x 2 col-tiles
// (8 WMMA per 4 global-B + 8 LDS-A loads: math-issue balanced).

__global__ __launch_bounds__(256) void gemm_bias_relu_bf16(
    const u16* __restrict__ A, const u16* __restrict__ BT,
    const float* __restrict__ bias, u16* __restrict__ C,
    int M, int N, int K)
{
    __shared__ u16 As[128 * 72];          // 128 rows x KC=64 (+8 pad: conflict-free)
    const int tid  = threadIdx.x;
    const int wave = tid >> 5, lane = tid & 31;
    const int half = lane >> 4, lr = lane & 15;
    const int wm = wave & 1;              // row group: 4 tiles (64 rows)
    const int wn = wave >> 1;             // col group: 2 tiles (32 cols)
    const int row0 = blockIdx.x * 128;
    const int col0 = blockIdx.y * 128;

    v8f acc[4][2];
#pragma unroll
    for (int i = 0; i < 4; ++i)
#pragma unroll
        for (int j = 0; j < 2; ++j) acc[i][j] = (v8f){0,0,0,0,0,0,0,0};

    for (int kc = 0; kc < K; kc += 64) {
        __syncthreads();
#pragma unroll
        for (int i = 0; i < 4; ++i) {     // 1024 16B-chunks: 128 rows x 8 chunks
            int c  = tid + i * 256;
            int r  = c >> 3;
            int k8 = (c & 7) * 8;
            copy16_g2l(&A[(size_t)(row0 + r) * K + kc + k8], &As[r * 72 + k8]);
        }
        staging_fence();
        __syncthreads();
#pragma unroll
        for (int kk = 0; kk < 64; kk += 32) {
            v16bf af[4];
#pragma unroll
            for (int i = 0; i < 4; ++i)
                af[i] = make_frag(&As[(wm * 64 + i * 16 + lr) * 72 + kk + 8 * half]);
#pragma unroll
            for (int j = 0; j < 2; ++j) {
                int n = col0 + (wn * 2 + j) * 16 + lr;
                v16bf bf = make_frag(&BT[(size_t)n * K + kc + kk + 8 * half]);
#pragma unroll
                for (int i = 0; i < 4; ++i)
                    acc[i][j] = __builtin_amdgcn_wmma_f32_16x16x32_bf16(
                        false, af[i], false, bf, (short)0, acc[i][j], false, false);
            }
        }
    }
    // epilogue: bias + relu + bf16 store. C layout: VGPR v -> M = v + 8*half, N = lr.
#pragma unroll
    for (int j = 0; j < 2; ++j) {
        int col = col0 + (wn * 2 + j) * 16 + lr;
        float bv = bias[col];
#pragma unroll
        for (int i = 0; i < 4; ++i)
#pragma unroll
            for (int v = 0; v < 8; ++v) {
                int row = row0 + wm * 64 + i * 16 + v + 8 * half;
                float x = acc[i][j][v] + bv;
                x = x > 0.f ? x : 0.f;
                C[(size_t)row * N + col] = f32_to_bf16(x);
            }
    }
}

// ---------------- fused GEMM3 + rational-quadratic spline ------------------
// H [16384 x 1024] bf16 (hdn2), W2T [12032 x 1024] bf16, b2 f32[12032],
// x2 f32[16384 x 256] -> y f32[16384 x 256], logdet f32[16384].
// Block: 64 rows, full K panel in LDS (read hdn once from HBM), sweep all
// 256 spline groups. 8 waves = 2 row-pairs x 4 group-phases; each wave:
// 2 M-tiles x 3 N-tiles per group (6 WMMA per 6 global-B loads per K-step).

__global__ __launch_bounds__(256) void gemm3_rqs_fused(
    const u16*  __restrict__ H,
    const u16*  __restrict__ W2T,
    const float* __restrict__ b2,
    const float* __restrict__ x2,
    float* __restrict__ y_out,
    float* __restrict__ ld_out)
{
    __shared__ u16   Hs[64 * 1032];       // 64 x 1024 (+8 pad) = 132,096 B
    __shared__ float stage[8][16][48];    // per-wave p-tile        24,576 B
    __shared__ float ldred[8][2][16];     // logdet partials          1,024 B

    const int tid  = threadIdx.x;
    const int wave = tid >> 5, lane = tid & 31;
    const int half = lane >> 4, lr = lane & 15;
    const int mh   = wave & 1;            // row-tiles {2mh, 2mh+1}
    const int ph   = wave >> 1;           // group phase 0..3
    const int row0 = blockIdx.x * 64;

    // Stage the 64x1024 activation panel once (async DMA to LDS if available).
#pragma unroll 4
    for (int i = 0; i < 32; ++i) {
        int c  = tid + i * 256;           // 8192 16B-chunks
        int r  = c >> 7;
        int k8 = (c & 127) * 8;
        copy16_g2l(&H[(size_t)(row0 + r) * 1024 + k8], &Hs[r * 1032 + k8]);
    }
    staging_fence();
    __syncthreads();

    float ldacc[2] = {0.f, 0.f};

    for (int t = 0; t < 64; ++t) {
        const int g = ph + 4 * t;         // d2 index 0..255

        // Warm WGP$ with next group's W2T rows while this group's WMMAs run.
        if (t + 1 < 64) {
#pragma unroll
            for (int jt = 0; jt < 3; ++jt) {
                int n = 47 * (g + 4) + jt * 16 + lr;
                n = n > 12031 ? 12031 : n;
                __builtin_prefetch(&W2T[(size_t)n * 1024], 0, 1);
            }
        }

        v8f acc[2][3];
#pragma unroll
        for (int i = 0; i < 2; ++i)
#pragma unroll
            for (int jt = 0; jt < 3; ++jt) acc[i][jt] = (v8f){0,0,0,0,0,0,0,0};

#pragma unroll 4
        for (int k0 = 0; k0 < 1024; k0 += 32) {
            v16bf af[2];
#pragma unroll
            for (int i = 0; i < 2; ++i)
                af[i] = make_frag(&Hs[((2 * mh + i) * 16 + lr) * 1032 + k0 + 8 * half]);
#pragma unroll
            for (int jt = 0; jt < 3; ++jt) {
                int n = 47 * g + jt * 16 + lr;          // col 47 of tile2 = junk
                n = n > 12031 ? 12031 : n;              // clamp (ignored col)
                v16bf bf = make_frag(&W2T[(size_t)n * 1024 + k0 + 8 * half]);
#pragma unroll
                for (int i = 0; i < 2; ++i)
                    acc[i][jt] = __builtin_amdgcn_wmma_f32_16x16x32_bf16(
                        false, af[i], false, bf, (short)0, acc[i][jt], false, false);
            }
        }

        // Two spline passes (one per M-tile) through the per-wave stage.
        // Same-wave LDS is in-order: pass-1 writes cannot pass pass-0 reads.
#pragma unroll
        for (int pass = 0; pass < 2; ++pass) {
#pragma unroll
            for (int jt = 0; jt < 3; ++jt) {
                int c = jt * 16 + lr;
                float bv = (c < 47) ? b2[47 * g + c] : 0.f;
#pragma unroll
                for (int v = 0; v < 8; ++v)
                    stage[wave][v + 8 * half][c] = acc[pass][jt][v] + bv;
            }

            if (lane < 16) {
                const int rloc = (2 * mh + pass) * 16 + lane;   // row in block
                float pv[48];
                const float4* p4 = (const float4*)stage[wave][lane];
#pragma unroll
                for (int i = 0; i < 12; ++i) {
                    float4 q = p4[i];
                    pv[4*i] = q.x; pv[4*i+1] = q.y; pv[4*i+2] = q.z; pv[4*i+3] = q.w;
                }
                float w_[16], h_[16], d_[17];
                // widths softmax -> bin widths (scaled by 2*TAIL)
                float m = pv[0];
#pragma unroll
                for (int i = 1; i < 16; ++i) m = fmaxf(m, pv[i]);
                float s = 0.f;
#pragma unroll
                for (int i = 0; i < 16; ++i) { float e = __expf(pv[i] - m); w_[i] = e; s += e; }
                float inv = 1.0f / s;
#pragma unroll
                for (int i = 0; i < 16; ++i)
                    w_[i] = (2.0f * TAILC) * (MIN_VAL + (1.0f - MIN_TOTAL) * w_[i] * inv);
                // heights softmax
                m = pv[16];
#pragma unroll
                for (int i = 1; i < 16; ++i) m = fmaxf(m, pv[16 + i]);
                s = 0.f;
#pragma unroll
                for (int i = 0; i < 16; ++i) { float e = __expf(pv[16 + i] - m); h_[i] = e; s += e; }
                inv = 1.0f / s;
#pragma unroll
                for (int i = 0; i < 16; ++i)
                    h_[i] = (2.0f * TAILC) * (MIN_VAL + (1.0f - MIN_TOTAL) * h_[i] * inv);
                // derivatives: pads are exactly 1.0 (MIN_VAL + softplus(TAIL_CONST))
                d_[0] = 1.0f; d_[16] = 1.0f;
#pragma unroll
                for (int i = 0; i < 15; ++i) {
                    float u = pv[32 + i];
                    float sp = (u > 15.f) ? u : __logf(1.f + __expf(u));
                    d_[i + 1] = MIN_VAL + sp;
                }

                float x = x2[(size_t)(row0 + rloc) * 256 + g];
                bool inside = (x > -TAILC) && (x < TAILC);
                float xc = fminf(fmaxf(x, -TAILC), TAILC);

                // predicated bin select (15 unrolled steps, all regs)
                float cw_lo = -TAILC, ch_lo = -TAILC;
                float run = -TAILC, chrun = -TAILC;
                float wk = w_[0], hk = h_[0], dk = d_[0], dk1 = d_[1];
#pragma unroll
                for (int i = 0; i < 15; ++i) {
                    run += w_[i]; chrun += h_[i];
                    if (xc >= run) {
                        cw_lo = run; ch_lo = chrun;
                        wk = w_[i + 1]; hk = h_[i + 1];
                        dk = d_[i + 1]; dk1 = d_[i + 2];
                    }
                }
                float sk  = hk / wk;
                float th  = (xc - cw_lo) / wk;
                float om  = 1.f - th;
                float t1m = th * om;
                float den = sk + (dk + dk1 - 2.f * sk) * t1m;
                float y   = ch_lo + hk * (sk * th * th + dk * t1m) / den;
                float der = sk * sk * (dk1 * th * th + 2.f * sk * t1m + dk * om * om)
                            / (den * den);
                y_out[(size_t)(row0 + rloc) * 256 + g] = inside ? y : x;
                ldacc[pass] += inside ? __logf(der) : 0.f;
            }
        }
    }

    if (lane < 16) {
        ldred[wave][0][lane] = ldacc[0];
        ldred[wave][1][lane] = ldacc[1];
    }
    __syncthreads();
    if (tid < 64) {   // deterministic fixed-order reduction over the 4 phases
        int mh2 = tid >> 5, i2 = (tid >> 4) & 1, rr = tid & 15;
        float s = 0.f;
#pragma unroll
        for (int p = 0; p < 4; ++p) s += ldred[2 * p + mh2][i2][rr];
        ld_out[row0 + tid] = s;
    }
}

// ---------------------------------------------------------------------------

extern "C" void kernel_launch(void* const* d_in, const int* in_sizes, int n_in,
                              void* d_out, int out_size, void* d_ws, size_t ws_size,
                              hipStream_t stream) {
    const float* x1 = (const float*)d_in[0];
    const float* x2 = (const float*)d_in[1];
    const float* W0 = (const float*)d_in[2];
    const float* b0 = (const float*)d_in[3];
    const float* W1 = (const float*)d_in[4];
    const float* b1 = (const float*)d_in[5];
    const float* W2 = (const float*)d_in[6];
    const float* b2 = (const float*)d_in[7];

    char* ws = (char*)d_ws;                       // ~99 MB total
    u16* x1bf = (u16*)(ws);                       //  8,388,608 B
    u16* W0T  = (u16*)(ws + 8388608);             //    524,288 B
    u16* W1T  = (u16*)(ws + 8912896);             //  2,097,152 B
    u16* W2T  = (u16*)(ws + 11010048);            // 24,641,536 B
    u16* hdn1 = (u16*)(ws + 35651584);            // 33,554,432 B
    u16* hdn2 = (u16*)(ws + 69206016);            // 33,554,432 B

    float* y  = (float*)d_out;
    float* ld = (float*)d_out + (size_t)16384 * 256;

    cvt_f32_bf16<<<16384, 256, 0, stream>>>(x1, x1bf, 16384 * 256);
    transpose_cvt<<<(1024 * 256) / 256, 256, 0, stream>>>(W0, W0T, 256, 1024);
    transpose_cvt<<<(1024 * 1024) / 256, 256, 0, stream>>>(W1, W1T, 1024, 1024);
    transpose_cvt<<<(12032 * 1024) / 256, 256, 0, stream>>>(W2, W2T, 1024, 12032);

    gemm_bias_relu_bf16<<<dim3(128, 8), 256, 0, stream>>>(x1bf, W0T, b0, hdn1,
                                                          16384, 1024, 256);
    gemm_bias_relu_bf16<<<dim3(128, 8), 256, 0, stream>>>(hdn1, W1T, b1, hdn2,
                                                          16384, 1024, 1024);
    gemm3_rqs_fused<<<256, 256, 0, stream>>>(hdn2, W2T, b2, x2, y, ld);
}
